// KalmanNetSimple_79628693668162
// MI455X (gfx1250) — matrix operand order repaired and checked by reading, not compiled, not run on hardware
//
#include <hip/hip_runtime.h>
#include <hip/hip_bf16.h>
#include <math.h>

// ---- problem constants (match reference) ----
#define T_STEPS 2048
#define H1      2560
#define GDIM    5120
#define H2D     128
#define DTC     0.1f

typedef __attribute__((ext_vector_type(16))) __bf16   v16bf;
typedef __attribute__((ext_vector_type(8)))  float    v8f;
typedef __attribute__((ext_vector_type(4)))  unsigned u32x4;

union TileU { struct { u32x4 lo, hi; } s; v16bf v; };

struct Params {
  // inputs (fp32)
  const float *obs, *init_state, *A, *Hm;
  const float *W1, *b1, *b_ih, *b_hh, *W2, *b2, *W3, *b3, *h0;
  // workspace
  __bf16 *Wih_b, *Whh_b;       // bf16 copies of the big weights
  float  *hF;                  // fp32 GRU hidden state [G]
  __bf16 *hB0, *hB1;           // bf16 hidden, double-buffered
  __bf16 *x1b;                 // bf16 x1 [H1]
  __bf16 *zbuf;                // 64B of zeros (B-operand for non-carrier lanes)
  float  *x2_acc;              // [128] fp32 atomic accumulator
  float  *prior_ws, *innov_ws; // [16] each
  unsigned *cnt;               // tail-block counter
  float  *out;                 // estimates [T,16]
};

// ---------- fp32 -> bf16 weight conversion (once per launch) ----------
__global__ void cvt_bf16(const float* __restrict__ s, __bf16* __restrict__ d, size_t n) {
  size_t i  = (size_t)blockIdx.x * blockDim.x + threadIdx.x;
  size_t st = (size_t)gridDim.x * blockDim.x;
  for (; i < n; i += st) d[i] = (__bf16)s[i];
}

// ---------- init: state + step-0 pre-stage (prior, innov, feat, x1) ----------
__global__ void kn_init(Params p) {
  __shared__ float sh_prior[16], sh_innov[16], sh_feat[32];
  const int tid = threadIdx.x;
  for (int g = tid; g < GDIM; g += 256) { float v = p.h0[g]; p.hF[g] = v; p.hB0[g] = (__bf16)v; }
  if (tid < 128) p.x2_acc[tid] = 0.f;
  if (tid < 32)  p.zbuf[tid] = (__bf16)0.f;
  if (tid == 0)  *p.cnt = 0u;
  if (tid < 16) {
    float s = 0.f;
    for (int j = 0; j < 16; ++j) s += p.A[tid*16 + j] * p.init_state[j];
    float pr = p.init_state[tid] + DTC * s;          // prior = (I + dt*A) @ post
    sh_prior[tid] = pr; p.prior_ws[tid] = pr;
  }
  __syncthreads();
  if (tid < 16) {
    float s = 0.f;
    for (int m = 0; m < 16; ++m) s += p.Hm[tid*16 + m] * sh_prior[m];
    float inn = p.obs[tid] - s;
    sh_innov[tid] = inn; p.innov_ws[tid] = inn;
  }
  __syncthreads();
  if (tid == 0) {
    float si = 0.f;
    for (int k = 0; k < 16; ++k) si += sh_innov[k]*sh_innov[k];
    float di = fmaxf(sqrtf(si), 1e-12f);
    for (int k = 0; k < 16; ++k) { sh_feat[k] = sh_innov[k]/di; sh_feat[16+k] = 0.f; } // fdiff==0
  }
  __syncthreads();
  for (int j = tid; j < H1; j += 256) {
    float s = p.b1[j];
    const float* w = p.W1 + (size_t)j * 32;
    for (int k = 0; k < 32; ++k) s += w[k] * sh_feat[k];
    p.x1b[j] = (__bf16)(s > 0.f ? s : 0.f);
  }
}

// ---------- one recurrence step ----------
__global__ void __launch_bounds__(256) kn_step(Params p, int t, int parity) {
  __shared__ float red[6][8][16];      // per-wave column-0 partials, 6 gate matrices
  __shared__ float hnew_s[16];
  __shared__ int   last_s;
  __shared__ float x2s[128], Kmat[256];
  __shared__ float pold[16], innv[16], pnew[16], prnew[16], fd[16], feat[32];

  const int tid  = threadIdx.x;
  const int wid  = tid >> 5, lane = tid & 31;
  const int row  = lane & 15, half = lane >> 4;
  const int g0   = blockIdx.x * 16;              // this block's 16-gate tile
  const __bf16* hb_r = parity ? p.hB1 : p.hB0;
  __bf16*       hb_w = parity ? p.hB0 : p.hB1;

  // Only lanes 0 and 16 (row==0) carry real B data (N=0 column); all other
  // lanes read a pinned 64B zero buffer (stride 0 -> always L0-hot).
  const bool carrier = (row == 0);

  // ---- six matvec tiles via WMMA bf16 (vector in column 0 of B) ----
#pragma unroll
  for (int m = 0; m < 6; ++m) {
    const int     Kd = (m < 3) ? H1 : GDIM;
    const int     nC = Kd / 32;                   // 80 or 160 K-chunks
    const int     per = nC / 8;                   // 10 or 20 chunks per wave (even)
    const __bf16* Wb = (m < 3) ? p.Wih_b : p.Whh_b;
    const __bf16* xv = (m < 3) ? p.x1b   : hb_r;
    const int   mrow = (m < 3) ? m : (m - 3);

    const int c0 = wid * per;                     // contiguous K-range per wave
    const __bf16* ap = Wb + ((size_t)(mrow*GDIM + g0 + row)) * Kd + c0*32 + half*8;
    const __bf16* xp = carrier ? (xv + c0*32 + half*8) : p.zbuf;
    const int    xst = carrier ? 64 : 0;          // per pair of chunks

    v8f acc; for (int i = 0; i < 8; ++i) acc[i] = 0.f;
    for (int k = 0; k < per; k += 2) {            // hand-paired for load/WMMA overlap
      TileU a0, b0, a1, b1;
      a0.s.lo = *(const u32x4*)ap;                // chunk k:   K {0..7 | 8..15}
      a0.s.hi = *(const u32x4*)(ap + 16);         //            K {16..23 | 24..31}
      b0.s.lo = *(const u32x4*)xp;
      b0.s.hi = *(const u32x4*)(xp + 16);
      a1.s.lo = *(const u32x4*)(ap + 32);         // chunk k+1
      a1.s.hi = *(const u32x4*)(ap + 48);
      b1.s.lo = *(const u32x4*)(xp + (xst >> 1));
      b1.s.hi = *(const u32x4*)(xp + (xst >> 1) + 16);
      __builtin_prefetch(ap + 128, 0, 0);         // 256B ahead in this row stream
      acc = __builtin_amdgcn_wmma_f32_16x16x32_bf16(
          false, a0.v, false, b0.v, (short)0, acc, false, false);
      acc = __builtin_amdgcn_wmma_f32_16x16x32_bf16(
          false, a1.v, false, b1.v, (short)0, acc, false, false);
      ap += 64; xp += xst;
    }
    // D column 0 lives in lane 0 (rows 0..7) and lane 16 (rows 8..15)
    if (lane == 0 || lane == 16)
      for (int r8 = 0; r8 < 8; ++r8) red[m][wid][half*8 + r8] = acc[r8];
  }
  __syncthreads();

  // ---- GRU gates + h update for this tile ----
  if (tid < 16) {
    int g = g0 + tid;
    float s[6];
    for (int m = 0; m < 6; ++m) {
      float a = 0.f;
      for (int w = 0; w < 8; ++w) a += red[m][w][tid];
      s[m] = a;
    }
    float ir = s[0] + p.b_ih[g];
    float iz = s[1] + p.b_ih[GDIM + g];
    float in_ = s[2] + p.b_ih[2*GDIM + g];
    float hr = s[3] + p.b_hh[g];
    float hz = s[4] + p.b_hh[GDIM + g];
    float hn = s[5] + p.b_hh[2*GDIM + g];
    float r  = 1.f / (1.f + expf(-(ir + hr)));
    float z  = 1.f / (1.f + expf(-(iz + hz)));
    float n  = tanhf(in_ + r * hn);
    float hnew = (1.f - z) * n + z * p.hF[g];
    p.hF[g]  = hnew;
    hb_w[g]  = (__bf16)hnew;
    hnew_s[tid] = hnew;
  }
  __syncthreads();

  // ---- W2 @ h_new partials (this tile's 16 columns) ----
  if (tid < 128) {
    const float* w = p.W2 + (size_t)tid * GDIM + g0;
    float s = 0.f;
    for (int i = 0; i < 16; ++i) s += w[i] * hnew_s[i];
    atomicAdd(&p.x2_acc[tid], s);
  }
  __threadfence();
  __syncthreads();
  if (tid == 0) {
    unsigned old = atomicAdd(p.cnt, 1u);
    last_s = (old == (unsigned)(gridDim.x - 1)) ? 1 : 0;
    if (last_s) *p.cnt = 0u;                       // reset for next step
  }
  __syncthreads();
  if (!last_s) return;
  __threadfence();                                 // acquire all blocks' writes

  // ---- tail block: x2 -> K -> posterior, then next step's pre-stage ----
  if (tid < 16)  { pold[tid] = p.prior_ws[tid]; innv[tid] = p.innov_ws[tid]; }
  if (tid < 128) {
    float v = p.x2_acc[tid] + p.b2[tid];
    x2s[tid] = v > 0.f ? v : 0.f;
    p.x2_acc[tid] = 0.f;                           // re-zero for next step
  }
  __syncthreads();
  {
    float s = p.b3[tid];
    const float* w = p.W3 + (size_t)tid * H2D;
    for (int j = 0; j < H2D; ++j) s += w[j] * x2s[j];
    Kmat[tid] = s;                                 // K (16x16) flat
  }
  __syncthreads();
  if (tid < 16) {
    float s = pold[tid];
    for (int n = 0; n < 16; ++n) s += Kmat[tid*16 + n] * innv[n];
    pnew[tid] = s;
    p.out[(size_t)t * 16 + tid] = s;               // estimates[t]
  }
  __syncthreads();
  if (t + 1 >= T_STEPS) return;

  if (tid < 16) {
    float s = 0.f;
    for (int j = 0; j < 16; ++j) s += p.A[tid*16 + j] * pnew[j];
    float pr = pnew[tid] + DTC * s;                // prior_{t+1}
    prnew[tid] = pr; p.prior_ws[tid] = pr;
    fd[tid] = pnew[tid] - pold[tid];               // fdiff_{t+1} = post_new - prior_t
  }
  __syncthreads();
  if (tid < 16) {
    float s = 0.f;
    for (int m = 0; m < 16; ++m) s += p.Hm[tid*16 + m] * prnew[m];
    float inn = p.obs[(size_t)(t + 1) * 16 + tid] - s;
    innv[tid] = inn; p.innov_ws[tid] = inn;
  }
  __syncthreads();
  if (tid == 0) {
    float si = 0.f, sf = 0.f;
    for (int k = 0; k < 16; ++k) { si += innv[k]*innv[k]; sf += fd[k]*fd[k]; }
    float di = fmaxf(sqrtf(si), 1e-12f), df = fmaxf(sqrtf(sf), 1e-12f);
    for (int k = 0; k < 16; ++k) { feat[k] = innv[k]/di; feat[16+k] = fd[k]/df; }
  }
  __syncthreads();
  for (int j = tid; j < H1; j += 256) {            // x1_{t+1} = relu(W1@feat+b1)
    float s = p.b1[j];
    const float* w = p.W1 + (size_t)j * 32;
    for (int k = 0; k < 32; ++k) s += w[k] * feat[k];
    p.x1b[j] = (__bf16)(s > 0.f ? s : 0.f);
  }
}

extern "C" void kernel_launch(void* const* d_in, const int* in_sizes, int n_in,
                              void* d_out, int out_size, void* d_ws, size_t ws_size,
                              hipStream_t stream) {
  (void)in_sizes; (void)n_in; (void)out_size; (void)ws_size;
  Params p;
  p.obs        = (const float*)d_in[0];
  p.init_state = (const float*)d_in[1];
  p.A          = (const float*)d_in[2];
  p.Hm         = (const float*)d_in[3];
  p.W1         = (const float*)d_in[4];
  p.b1         = (const float*)d_in[5];
  const float* W_ih = (const float*)d_in[6];
  p.b_ih       = (const float*)d_in[7];
  const float* W_hh = (const float*)d_in[8];
  p.b_hh       = (const float*)d_in[9];
  p.W2         = (const float*)d_in[10];
  p.b2         = (const float*)d_in[11];
  p.W3         = (const float*)d_in[12];
  p.b3         = (const float*)d_in[13];
  p.h0         = (const float*)d_in[14];
  p.out        = (float*)d_out;

  char* ws = (char*)d_ws;
  size_t off = 0;
  auto take = [&](size_t bytes) { size_t o = off; off += (bytes + 255) & ~(size_t)255; return o; };
  const size_t nIH = (size_t)3 * GDIM * H1;        // 39,321,600
  const size_t nHH = (size_t)3 * GDIM * GDIM;      // 78,643,200
  p.Wih_b    = (__bf16*)  (ws + take(nIH * 2));    // ~236 MB total bf16 weights
  p.Whh_b    = (__bf16*)  (ws + take(nHH * 2));
  p.hF       = (float*)   (ws + take(GDIM * 4));
  p.hB0      = (__bf16*)  (ws + take(GDIM * 2));
  p.hB1      = (__bf16*)  (ws + take(GDIM * 2));
  p.x1b      = (__bf16*)  (ws + take(H1 * 2));
  p.zbuf     = (__bf16*)  (ws + take(64));
  p.x2_acc   = (float*)   (ws + take(128 * 4));
  p.prior_ws = (float*)   (ws + take(16 * 4));
  p.innov_ws = (float*)   (ws + take(16 * 4));
  p.cnt      = (unsigned*)(ws + take(64));

  cvt_bf16<<<2048, 256, 0, stream>>>(W_ih, p.Wih_b, nIH);
  cvt_bf16<<<2048, 256, 0, stream>>>(W_hh, p.Whh_b, nHH);
  kn_init<<<1, 256, 0, stream>>>(p);
  for (int t = 0; t < T_STEPS; ++t)
    kn_step<<<320, 256, 0, stream>>>(p, t, t & 1);
}
